// EncoderLayer_54692113547879
// MI455X (gfx1250) — matrix operand order repaired and checked by reading, not compiled, not run on hardware
//
#include <hip/hip_runtime.h>
#include <hip/hip_bf16.h>

// Problem constants (match reference)
#define Bc 4
#define Nc_ 2048
#define Kc 32
#define Hc 128
#define NIc 256
#define DFFc 512

typedef __attribute__((ext_vector_type(16))) _Float16 v16h;
typedef __attribute__((ext_vector_type(4)))  _Float16 v4h;
typedef __attribute__((ext_vector_type(8)))  float    v8f;

#define NEGF (-3.4028234663852886e38f)

__device__ __forceinline__ float gelu_exact(float x) {
  return 0.5f * x * (1.0f + erff(x * 0.7071067811865475f));
}

// ======================= Block-tiled, double-buffered WMMA GEMM =======================
// C = act(A @ W + bias).  A: [M,Ka] f32, W: [Ka,Nc] f32, C: [M,Nc] f32.
// Block: 256 threads = 8 waves; block tile 128x128; k-step 32; 2-deep LDS pipeline.
// A/W tiles staged in LDS as f16, pre-swizzled to the WMMA fragment layout so each
// lane's fragment is one contiguous 32B LDS read (2x ds_load_b128).
// Requires: M % 128 == 0, Nc % 128 == 0, Ka % 32 == 0 (all call sites satisfy).
__global__ __launch_bounds__(256)
void gemm_wmma_kernel(const float* __restrict__ A, const float* __restrict__ W,
                      const float* __restrict__ bias, float* __restrict__ C,
                      int M, int Ka, int Nc, int act)
{
  const int gn = Nc >> 7;                       // n-blocks
  const int bm = blockIdx.x / gn;
  const int bn = blockIdx.x - bm * gn;
  const int blockM = bm << 7;
  const int blockN = bn << 7;

  const int t    = threadIdx.x;
  const int wv_  = t >> 5;                      // wave id = mtile (0..7)
  const int lane = t & 31;
  const int lhi  = lane >> 4;
  const int lm   = lane & 15;

  // [buf][tile*32 + lane] -> one 16-half fragment slice per lane (32B each)
  __shared__ v16h AsT[2][256];                  // 2 x 8 mtiles x 32 lanes (8KB each)
  __shared__ v16h WsT[2][256];                  // 2 x 8 ntiles x 32 lanes

  // Per-thread tile-load geometry (constant across k-steps):
  //  A tile 128x32: thread covers rows arow0+32j, 4 consecutive k at akk.
  //  W tile 32x128: thread covers k rows wkk0+8j, 4 consecutive n at wn4.
  const int arow0 = t >> 3;
  const int akk   = (t & 7) << 2;
  const int wkk0  = t >> 5;
  const int wn4   = (t & 31) << 2;

  float4 ar[4], wr[4];

  auto load_tile = [&](int k0) {
    #pragma unroll
    for (int j = 0; j < 4; ++j)
      ar[j] = *reinterpret_cast<const float4*>(
          &A[(size_t)(blockM + arow0 + 32 * j) * (size_t)Ka + k0 + akk]);
    #pragma unroll
    for (int j = 0; j < 4; ++j)
      wr[j] = *reinterpret_cast<const float4*>(
          &W[(size_t)(k0 + wkk0 + 8 * j) * (size_t)Nc + blockN + wn4]);
    if (k0 + 32 < Ka) {  // prefetch one more k-step ahead
      __builtin_prefetch(&A[(size_t)(blockM + arow0) * (size_t)Ka + k0 + 32 + akk], 0, 1);
      __builtin_prefetch(&W[(size_t)(k0 + 32 + wkk0) * (size_t)Nc + blockN + wn4], 0, 1);
    }
  };

  auto store_tile = [&](int buf) {
    _Float16* asp = reinterpret_cast<_Float16*>(AsT[buf]);
    _Float16* wsp = reinterpret_cast<_Float16*>(WsT[buf]);
    // swizzle (m,kk) -> lane L = m + 16*((kk>>3)&1), half i = (kk&7) + 8*(kk>>4)
    #pragma unroll
    for (int j = 0; j < 4; ++j) {
      const int r  = arow0 + 32 * j;
      const int mt = r >> 4, m = r & 15;
      const int L  = m + (((akk >> 3) & 1) << 4);
      const int i0 = (akk & 7) + ((akk >> 4) << 3);
      v4h hv; hv[0] = (_Float16)ar[j].x; hv[1] = (_Float16)ar[j].y;
              hv[2] = (_Float16)ar[j].z; hv[3] = (_Float16)ar[j].w;
      *reinterpret_cast<v4h*>(&asp[((mt << 5) + L) * 16 + i0]) = hv;
    }
    // swizzle (kk,n) -> lane L = (n&15) + 16*((kk>>3)&1), half i = (kk&7) + 8*(kk>>4)
    #pragma unroll
    for (int j = 0; j < 4; ++j) {
      const int kk   = wkk0 + 8 * j;
      const int lhiw = (kk >> 3) & 1;
      const int iw   = (kk & 7) + ((kk >> 4) << 3);
      const float vals[4] = {wr[j].x, wr[j].y, wr[j].z, wr[j].w};
      #pragma unroll
      for (int u = 0; u < 4; ++u) {
        const int n  = wn4 + u;
        const int nt = n >> 4;
        const int L  = (n & 15) + (lhiw << 4);
        wsp[((nt << 5) + L) * 16 + iw] = (_Float16)vals[u];
      }
    }
  };

  v8f acc[8];
  #pragma unroll
  for (int i = 0; i < 8; ++i) acc[i] = v8f{0.f,0.f,0.f,0.f,0.f,0.f,0.f,0.f};

  // ---- pipeline prologue ----
  load_tile(0);
  store_tile(0);
  __syncthreads();

  int cur = 0;
  for (int k0 = 0; k0 < Ka; k0 += 32) {
    const bool hasNext = (k0 + 32) < Ka;
    if (hasNext) load_tile(k0 + 32);            // overlap global loads with compute

    const v16h a = AsT[cur][(wv_ << 5) + lane];
    #pragma unroll
    for (int tn2 = 0; tn2 < 8; ++tn2) {
      const v16h b = WsT[cur][(tn2 << 5) + lane];
      acc[tn2] = __builtin_amdgcn_wmma_f32_16x16x32_f16(false, a, false, b,
                                                        (short)0, acc[tn2], false, false);
    }

    if (hasNext) {
      store_tile(cur ^ 1);
      __syncthreads();
      cur ^= 1;
    }
  }

  // ---- epilogue ----
  #pragma unroll
  for (int tn2 = 0; tn2 < 8; ++tn2) {
    const int col = blockN + (tn2 << 4) + lm;
    const float bv = bias ? bias[col] : 0.0f;
    #pragma unroll
    for (int r = 0; r < 8; ++r) {
      const int row = blockM + (wv_ << 4) + (lhi << 3) + r;
      float v = acc[tn2][r] + bv;
      if (act == 1) v = gelu_exact(v);
      C[(size_t)row * (size_t)Nc + col] = v;
    }
  }
}

// ======================= Neighbor attention core =======================
__global__ __launch_bounds__(128)
void na_attn_kernel(const float* __restrict__ Q, const float* __restrict__ Kb,
                    const float* __restrict__ Vb, const float* __restrict__ mask,
                    float* __restrict__ Out)
{
  const int bn   = blockIdx.x;
  const int h    = threadIdx.x >> 5;
  const int lane = threadIdx.x & 31;
  __shared__ float qs[Hc];
  qs[threadIdx.x] = Q[(size_t)bn * Hc + threadIdx.x];
  __syncthreads();

  const float* Krow = Kb + ((size_t)bn * Kc + lane) * Hc + h * 32;
  float dot = 0.f;
  #pragma unroll
  for (int d = 0; d < 32; ++d) dot += qs[h * 32 + d] * Krow[d];
  const float m = mask[(size_t)bn * Kc + lane];
  float logit = (m > 0.f) ? dot * 0.17677669529663687f : NEGF;   // /sqrt(32)

  float mx = logit;
  #pragma unroll
  for (int off = 16; off; off >>= 1) mx = fmaxf(mx, __shfl_xor(mx, off, 32));
  float e = expf(logit - mx);
  float s = e;
  #pragma unroll
  for (int off = 16; off; off >>= 1) s += __shfl_xor(s, off, 32);
  const float attend = m * (e / s);                              // attend = mask * softmax

  const float* Vh = Vb + (size_t)bn * Kc * Hc + h * 32 + lane;   // lane = d now
  float accv = 0.f;
  #pragma unroll
  for (int k = 0; k < 32; ++k) {
    float ak = __shfl(attend, k, 32);
    accv += ak * Vh[(size_t)k * Hc];
  }
  Out[(size_t)bn * Hc + h * 32 + lane] = accv;
}

// ======================= Residual + GELU + LayerNorm =======================
// out = LN(X + gelu(DH)) * rowmask ; DH==null -> plain LN(X); rowmask==null -> no mask.
__global__ __launch_bounds__(128)
void res_gelu_ln_kernel(const float* __restrict__ X, const float* __restrict__ DH,
                        const float* __restrict__ g, const float* __restrict__ b,
                        const float* __restrict__ rowmask, float* __restrict__ Out)
{
  const int row = blockIdx.x;
  const int h   = threadIdx.x;
  float v = X[(size_t)row * Hc + h];
  if (DH) v += gelu_exact(DH[(size_t)row * Hc + h]);

  __shared__ float red[4];
  float s = v;
  #pragma unroll
  for (int off = 16; off; off >>= 1) s += __shfl_xor(s, off, 32);
  if ((h & 31) == 0) red[h >> 5] = s;
  __syncthreads();
  const float mu = (red[0] + red[1] + red[2] + red[3]) * (1.0f / 128.0f);
  const float d  = v - mu;
  __syncthreads();
  float s2 = d * d;
  #pragma unroll
  for (int off = 16; off; off >>= 1) s2 += __shfl_xor(s2, off, 32);
  if ((h & 31) == 0) red[h >> 5] = s2;
  __syncthreads();
  const float var = (red[0] + red[1] + red[2] + red[3]) * (1.0f / 128.0f);
  float o = g[h] * d * (1.0f / sqrtf(var + 1e-5f)) + b[h];
  if (rowmask) o *= rowmask[row];
  Out[(size_t)row * Hc + h] = o;
}

// ======================= Outer product gather =======================
__global__ __launch_bounds__(128)
void outer_gather_kernel(const float* __restrict__ Xp, const int* __restrict__ Eidx,
                         float* __restrict__ Outer)
{
  const int bnk = blockIdx.x;            // (b*N+n)*K + k
  const int bn  = bnk >> 5;
  const int bb  = bn / Nc_;
  const int h   = threadIdx.x;
  const int idx = Eidx[bnk];
  float a = Xp[(size_t)bn * Hc + h];
  float c = Xp[((size_t)bb * Nc_ + idx) * Hc + h];
  Outer[(size_t)bnk * Hc + h] = a * c;
}

// ======================= Elementwise add =======================
__global__ __launch_bounds__(256)
void add2_kernel(const float* __restrict__ a, const float* __restrict__ b,
                 float* __restrict__ c, size_t n)
{
  size_t i = (size_t)blockIdx.x * 256 + threadIdx.x;
  if (i < n) c[i] = a[i] + b[i];
}

// ======================= Edge self-attention core =======================
// Faithful to the reference's raw .view reshapes (flat-index remaps verified):
//   logits2[i,j,k2] = L[k2>>3][(k2&7)*4 + (j>>3)][(j*4+i)&31] / sqrt(32)
__global__ __launch_bounds__(128)
void edge_attn_kernel(const float* __restrict__ EQ, const float* __restrict__ EK,
                      const float* __restrict__ EV, const float* __restrict__ mask,
                      float* __restrict__ AOut)
{
  const int bn = blockIdx.x;
  __shared__ float Qs[4096], Ks[4096], Vs[4096], Ls[4096];
  const size_t base = (size_t)bn * 4096;
  for (int i = threadIdx.x; i < 4096; i += 128) {
    Qs[i] = EQ[base + i];
    Ks[i] = EK[base + i];
    Vs[i] = EV[base + i];
  }
  __syncthreads();

  for (int e = threadIdx.x; e < 4096; e += 128) {
    const int nh = e >> 10, q = (e >> 5) & 31, kk = e & 31;
    const float* qp = Qs + (nh << 10) + (q << 5);
    const float* kp = Ks + (nh << 10) + (kk << 5);
    float dsum = 0.f;
    #pragma unroll
    for (int d = 0; d < 32; ++d) dsum += qp[d] * kp[d];
    Ls[e] = dsum;
  }
  __syncthreads();

  const int t = threadIdx.x;
  const int i = t >> 5;
  const int j = t & 31;
  const float mq = mask[(size_t)bn * Kc + j];
  const float inv = 0.17677669529663687f;
  float att[32];
  float mx = NEGF;
  #pragma unroll
  for (int k2 = 0; k2 < 32; ++k2) {
    const float mk = mask[(size_t)bn * Kc + k2];
    float lv = Ls[((k2 >> 3) << 10) + ((((k2 & 7) << 2) + (j >> 3)) << 5) + ((j * 4 + i) & 31)] * inv;
    lv = (mq > 0.f && mk > 0.f) ? lv : NEGF;
    att[k2] = lv;
    mx = fmaxf(mx, lv);
  }
  float ssum = 0.f;
  #pragma unroll
  for (int k2 = 0; k2 < 32; ++k2) { float ev = expf(att[k2] - mx); att[k2] = ev; ssum += ev; }
  const float rs = 1.0f / ssum;

  float o[32];
  #pragma unroll
  for (int d = 0; d < 32; ++d) o[d] = 0.f;
  for (int k2 = 0; k2 < 32; ++k2) {
    const float a = att[k2] * rs;
    const float* vp = Vs + (i << 10) + (k2 << 5);
    #pragma unroll
    for (int d = 0; d < 32; ++d) o[d] += a * vp[d];
  }
  float* op = AOut + ((size_t)bn * Kc + j) * Hc + (i << 5);
  #pragma unroll
  for (int d = 0; d < 32; ++d) op[d] = o[d];
}

// ======================= Host-side orchestration =======================
static inline void launch_gemm(const float* A, const float* W, const float* bias,
                               float* C, int M, int Ka, int Nc, int act, hipStream_t s)
{
  int blocks = (M / 128) * (Nc / 128);
  gemm_wmma_kernel<<<blocks, 256, 0, s>>>(A, W, bias, C, M, Ka, Nc, act);
}

extern "C" void kernel_launch(void* const* d_in, const int* in_sizes, int n_in,
                              void* d_out, int out_size, void* d_ws, size_t ws_size,
                              hipStream_t stream)
{
  (void)in_sizes; (void)n_in; (void)out_size; (void)ws_size;
  const float* h_V      = (const float*)d_in[0];
  const float* h_E      = (const float*)d_in[1];
  const float* h_EV     = (const float*)d_in[2];
  const float* mask_V   = (const float*)d_in[3];
  const float* mask_att = (const float*)d_in[4];
  const float* na_WQ = (const float*)d_in[5];
  const float* na_WK = (const float*)d_in[6];
  const float* na_WV = (const float*)d_in[7];
  const float* na_WO = (const float*)d_in[8];
  const float* ea_WK = (const float*)d_in[9];
  const float* ea_WV = (const float*)d_in[10];
  const float* ea_WO = (const float*)d_in[11];
  const float* d1_Win  = (const float*)d_in[12];
  const float* d1_bin  = (const float*)d_in[13];
  const float* d1_Wout = (const float*)d_in[14];
  const float* d1_bout = (const float*)d_in[15];
  const float* d2_Win  = (const float*)d_in[16];
  const float* d2_bin  = (const float*)d_in[17];
  const float* d2_Wout = (const float*)d_in[18];
  const float* d2_bout = (const float*)d_in[19];
  const float* op_W1   = (const float*)d_in[20];
  const float* op_b1   = (const float*)d_in[21];
  const float* op_W2   = (const float*)d_in[22];
  const float* op_b2   = (const float*)d_in[23];
  const float* op_g    = (const float*)d_in[24];
  const float* op_beta = (const float*)d_in[25];
  const float* n1_g = (const float*)d_in[26];
  const float* n1_b = (const float*)d_in[27];
  const float* n3_g = (const float*)d_in[28];
  const float* n3_b = (const float*)d_in[29];
  const float* n4_g = (const float*)d_in[30];
  const float* n4_b = (const float*)d_in[31];
  const int*   E_idx = (const int*)d_in[32];

  const int MV = Bc * Nc_;            // 8192 node rows
  const int ME = Bc * Nc_ * Kc;       // 262144 edge rows
  const size_t EH = (size_t)ME * Hc;  // 33,554,432 floats
  const size_t VH = (size_t)MV * Hc;  //  1,048,576 floats

  float* ws = (float*)d_ws;
  float* S0 = ws;                 // Kbuf -> outer -> EQ -> dhE raw
  float* S1 = S0 + EH;            // Vbuf -> hE1
  float* S2 = S1 + EH;            // oproj -> EK -> hE2
  float* S3 = S2 + EH;            // EV -> edge FFN hidden chunk
  float* S4 = S3 + EH;            // AOut -> dh2
  float* Qb   = S4 + EH;
  float* attO = Qb + VH;
  float* hv1  = attO + VH;
  float* tmpV = hv1 + VH;
  float* xln  = tmpV + VH;
  float* xpr  = xln + VH;
  float* t1   = xpr + VH;         // [8192, 512]

  float* outV = (float*)d_out;    // h_V result [B,N,H]
  float* outE = outV + VH;        // h_E result [B,N,K,H]

  // ---------- Stage 1: neighbor attention ----------
  launch_gemm(h_EV, na_WK, nullptr, S0, ME, NIc, Hc, 0, stream);   // K proj
  launch_gemm(h_EV, na_WV, nullptr, S1, ME, NIc, Hc, 0, stream);   // V proj
  launch_gemm(h_V,  na_WQ, nullptr, Qb, MV, Hc,  Hc, 0, stream);   // Q proj
  na_attn_kernel<<<MV, 128, 0, stream>>>(Qb, S0, S1, mask_att, attO);
  launch_gemm(attO, na_WO, nullptr, tmpV, MV, Hc, Hc, 0, stream);  // out proj
  res_gelu_ln_kernel<<<MV, 128, 0, stream>>>(h_V, tmpV, n1_g, n1_b, nullptr, hv1);

  // ---------- Stage 2: node FFN ----------
  launch_gemm(hv1, d1_Win,  d1_bin,  t1,   MV, Hc,   DFFc, 1, stream);
  launch_gemm(t1,  d1_Wout, d1_bout, tmpV, MV, DFFc, Hc,   0, stream);
  res_gelu_ln_kernel<<<MV, 128, 0, stream>>>(hv1, tmpV, n1_g, n1_b, mask_V, outV); // h_V output

  // ---------- Stage 3: outer product ----------
  res_gelu_ln_kernel<<<MV, 128, 0, stream>>>(outV, nullptr, op_g, op_beta, nullptr, xln);
  launch_gemm(xln, op_W1, op_b1, xpr, MV, Hc, Hc, 0, stream);
  outer_gather_kernel<<<ME, 128, 0, stream>>>(xpr, E_idx, S0);
  launch_gemm(S0, op_W2, op_b2, S2, ME, Hc, Hc, 0, stream);
  add2_kernel<<<(int)((EH + 255) / 256), 256, 0, stream>>>(h_E, S2, S1, EH); // hE1 in S1

  // ---------- Stage 4: edge self-attention ----------
  launch_gemm(S1, ea_WO, nullptr, S0, ME, Hc, Hc, 0, stream);  // "Q" via WO (faithful)
  launch_gemm(S1, ea_WK, nullptr, S2, ME, Hc, Hc, 0, stream);
  launch_gemm(S1, ea_WV, nullptr, S3, ME, Hc, Hc, 0, stream);
  edge_attn_kernel<<<MV, 128, 0, stream>>>(S0, S2, S3, mask_att, S4);
  launch_gemm(S4, ea_WO, nullptr, S0, ME, Hc, Hc, 0, stream);  // final out proj
  res_gelu_ln_kernel<<<ME, 128, 0, stream>>>(S1, S0, n3_g, n3_b, nullptr, S2); // hE2 in S2

  // ---------- Stage 5: edge FFN (4 row-chunks; hidden reuses one slab) ----------
  const int MC = ME / 4;
  for (int c = 0; c < 4; ++c) {
    const float* src = S2 + (size_t)c * MC * Hc;
    launch_gemm(src, d2_Win,  d2_bin,  S3,                       MC, Hc,   DFFc, 1, stream);
    launch_gemm(S3,  d2_Wout, d2_bout, S4 + (size_t)c * MC * Hc, MC, DFFc, Hc,   0, stream);
  }
  res_gelu_ln_kernel<<<ME, 128, 0, stream>>>(S2, S4, n4_g, n4_b, nullptr, outE); // h_E output
}
